// RNN_11553462026471
// MI455X (gfx1250) — matrix operand order repaired and checked by reading, not compiled, not run on hardware
//
#include <hip/hip_runtime.h>

// Elman RNN for MI455X (gfx1250, wave32, WMMA).
// B=128, T=2048, I=128, H=256, O=64.
//
// Plan:
//   Kernel 1 (parallel): pre[t][btile][ntile][frag] = f16(x_t @ W_xs + b_xs + b_ss)
//                        stored in WMMA-C fragment order -> recurrence loads it
//                        with a single b128 per lane per step.
//   Kernel 2 (serial over T): 8 persistent workgroups (one per 16-batch tile),
//                        16 waves each (one per 16-col tile of H). W_ss kept as
//                        f16 WMMA B-fragments in registers; h double-buffered in
//                        LDS (f16, padded stride for conflict-free ds_load_b128);
//                        8 chained v_wmma_f32_16x16x32_f16 per step, f32 tanh.
//   Fallback fused kernel if workspace < 128MB.

#define BB 128
#define TT 2048
#define II 128
#define HH 256
#define OO 64

typedef __attribute__((ext_vector_type(16))) _Float16 v16h;
typedef __attribute__((ext_vector_type(8)))  _Float16 v8h;
typedef __attribute__((ext_vector_type(8)))  float    v8f;
typedef __attribute__((ext_vector_type(4)))  float    v4f;

// padded LDS row strides (stride mod 64 dwords == 4 -> conflict-free b128 rows)
#define HP 264   // halves per h row in LDS (256 + 8)
#define XPH 136  // halves per x row in LDS (f16 path)
#define XPF 132  // floats per x row in LDS (f32 path)

static __device__ __forceinline__ float fast_tanh(float x) {
#if __has_builtin(__builtin_amdgcn_tanhf)
  return __builtin_amdgcn_tanhf(x);
#else
  float cx = fminf(fmaxf(x, -9.0f), 9.0f);
#if __has_builtin(__builtin_amdgcn_exp2f)
  float e = __builtin_amdgcn_exp2f(cx * 2.8853900817779268f); // exp(2x)
#else
  float e = exp2f(cx * 2.8853900817779268f);
#endif
  return (e - 1.0f) * __builtin_amdgcn_rcpf(e + 1.0f);
#endif
}

static __device__ __forceinline__ v16h cat8(v8h lo, v8h hi) {
  v16h a;
#pragma unroll
  for (int j = 0; j < 8; ++j) { a[j] = lo[j]; a[8 + j] = hi[j]; }
  return a;
}

// Load a 16x32 f16 A-fragment from an LDS row-major [16][stride] buffer,
// tile K origin = k0.  ISA 7.12.2: lane L holds row M=L&15; lanes<16 get
// K = k0+{0..7, 16..23}, lanes>=16 get K = k0+{8..15, 24..31}.
static __device__ __forceinline__ v16h load_a_f16(const _Float16* base, int stride,
                                                  int lane, int k0) {
  const int row = lane & 15;
  const int kc  = (lane >> 4) * 8;
  const _Float16* p = base + row * stride + k0 + kc;
  v8h lo = *(const v8h*)p;
  v8h hi = *(const v8h*)(p + 16);
  return cat8(lo, hi);
}

// Gather a 32x16 f16 B-fragment from a row-major f32 weight matrix W[K][ldw]
// at (k0, n0).  Layout: lane L -> column N = n0 + (L&15); elements 0..7 hold
// K = k0 + 8*(L>=16) + {0..7}, elements 8..15 hold K = k0 + 16 + 8*(L>=16) + {0..7}.
static __device__ __forceinline__ v16h load_b_w(const float* __restrict__ W, int ldw,
                                                int lane, int k0, int n0) {
  const int ncol = n0 + (lane & 15);
  const int kb   = k0 + (lane >> 4) * 8;
  v16h b;
#pragma unroll
  for (int j = 0; j < 8; ++j) {
    b[j]     = (_Float16)W[(kb + j) * ldw + ncol];
    b[8 + j] = (_Float16)W[(kb + 16 + j) * ldw + ncol];
  }
  return b;
}

#define WMMA_F16(a, b, c) \
  __builtin_amdgcn_wmma_f32_16x16x32_f16(false, (a), false, (b), (short)0, (c), false, false)

// ---------------------------------------------------------------------------
// Kernel 1: pre = f16(x @ W_xs + b_xs + b_ss), stored as WMMA-C fragments.
// grid (8, 256), block 256 (8 waves); each wg: one batch tile, 8 time steps;
// each wave: 2 n-tiles. W_xs fragments hoisted into registers.
// ---------------------------------------------------------------------------
__global__ __launch_bounds__(256) void rnn_pre_kernel(
    const float* __restrict__ x, const float* __restrict__ Wxs,
    const float* __restrict__ bxs, const float* __restrict__ bss,
    _Float16* __restrict__ pre) {
  __shared__ _Float16 xs[16 * XPH];

  const int tid = threadIdx.x;
  const int wave = tid >> 5, lane = tid & 31;
  const int btile = blockIdx.x;
  const int b0 = btile * 16;
  const int t0 = blockIdx.y * 8;
  const int col16 = lane & 15;

  v16h bx[4][2];
  float bias[2];
#pragma unroll
  for (int nn = 0; nn < 2; ++nn) {
    const int n0 = (2 * wave + nn) * 16;
#pragma unroll
    for (int kk = 0; kk < 4; ++kk) bx[kk][nn] = load_b_w(Wxs, HH, lane, kk * 32, n0);
    bias[nn] = bxs[n0 + col16] + bss[n0 + col16];
  }

  for (int ti = 0; ti < 8; ++ti) {
    const int t = t0 + ti;
    // cooperative x tile load: 16 rows x 128 f32 -> f16 LDS
    {
      const int r = tid >> 4;
      const int c0 = (tid & 15) * 8;
      const float* g = x + ((size_t)(b0 + r) * TT + t) * II + c0;
      v4f f0 = *(const v4f*)g;
      v4f f1 = *(const v4f*)(g + 4);
      v8h hv;
#pragma unroll
      for (int j = 0; j < 4; ++j) { hv[j] = (_Float16)f0[j]; hv[4 + j] = (_Float16)f1[j]; }
      *(v8h*)(&xs[r * XPH + c0]) = hv;
    }
    __syncthreads();

    v8f c0v, c1v;
#pragma unroll
    for (int r = 0; r < 8; ++r) { c0v[r] = bias[0]; c1v[r] = bias[1]; }
#pragma unroll
    for (int kk = 0; kk < 4; ++kk) {
      v16h a = load_a_f16(xs, XPH, lane, kk * 32);
      c0v = WMMA_F16(a, bx[kk][0], c0v);
      c1v = WMMA_F16(a, bx[kk][1], c1v);
    }
#pragma unroll
    for (int nn = 0; nn < 2; ++nn) {
      v8f cv = nn ? c1v : c0v;
      v8h hv;
#pragma unroll
      for (int j = 0; j < 8; ++j) hv[j] = (_Float16)cv[j];
      const size_t tile = ((size_t)t * 8 + btile) * 16 + (2 * wave + nn);
      *(v8h*)(pre + tile * 256 + (size_t)lane * 8) = hv;
    }
    __syncthreads();
  }
}

// shared epilogue: out = h_final @ W_out + b_out (waves 0..3)
static __device__ __forceinline__ void rnn_out_tail(
    const _Float16* hin, const float* __restrict__ Wout,
    const float* __restrict__ bout, float* __restrict__ out, int b0, int wave,
    int lane) {
  const int col16 = lane & 15, khalf = lane >> 4;
  const int ncol = wave * 16 + col16;
  v8f c;
  const float bb = bout[ncol];
#pragma unroll
  for (int r = 0; r < 8; ++r) c[r] = bb;
#pragma unroll
  for (int kk = 0; kk < 8; ++kk) {
    v16h a = load_a_f16(hin, HP, lane, kk * 32);
    v16h b = load_b_w(Wout, OO, lane, kk * 32, wave * 16);
    c = WMMA_F16(a, b, c);
  }
#pragma unroll
  for (int r = 0; r < 8; ++r)
    out[(size_t)(b0 + khalf * 8 + r) * OO + ncol] = c[r];
}

// ---------------------------------------------------------------------------
// Kernel 2: serial recurrence using precomputed `pre`.
// grid 8, block 512 (16 waves: wave w owns h columns [16w,16w+16)).
// ---------------------------------------------------------------------------
__global__ __launch_bounds__(512) void rnn_rec_kernel(
    const _Float16* __restrict__ pre, const float* __restrict__ Wss,
    const float* __restrict__ Wout, const float* __restrict__ bout,
    float* __restrict__ out) {
  __shared__ _Float16 hbuf[2][16 * HP];

  const int tid = threadIdx.x;
  const int wave = tid >> 5, lane = tid & 31;
  const int btile = blockIdx.x;
  const int b0 = btile * 16;
  const int col16 = lane & 15, khalf = lane >> 4;

  // W_ss fragments for this wave's n-tile: 8 K-tiles, resident for all T steps
  v16h bs[8];
#pragma unroll
  for (int kk = 0; kk < 8; ++kk) bs[kk] = load_b_w(Wss, HH, lane, kk * 32, wave * 16);

  // h0 = 0
  {
    unsigned* p = (unsigned*)&hbuf[0][0];
    for (int i = tid; i < 16 * HP / 2; i += 512) p[i] = 0u;
  }
  __syncthreads();

  // pre fragment base for (btile, wave); step stride = 8*16*256 halves
  const _Float16* preBase = pre + ((size_t)btile * 16 + wave) * 256 + (size_t)lane * 8;
  v8h pnext = *(const v8h*)preBase;  // t = 0 (software pipeline)

  for (int t = 0; t < TT; ++t) {
    const v8h pcur = pnext;
    if (t + 1 < TT) pnext = *(const v8h*)(preBase + (size_t)(t + 1) * 32768);
    if (t + 2 < TT) __builtin_prefetch(preBase + (size_t)(t + 2) * 32768, 0, 1);

    const _Float16* hin = &hbuf[t & 1][0];
    v8f c0 = {}, c1 = {};  // two chains for WMMA ILP
#pragma unroll
    for (int kk = 0; kk < 8; ++kk) {
      v16h a = load_a_f16(hin, HP, lane, kk * 32);
      if (kk & 1) c1 = WMMA_F16(a, bs[kk], c1);
      else        c0 = WMMA_F16(a, bs[kk], c0);
    }

    _Float16* hout = &hbuf[(t + 1) & 1][0];
#pragma unroll
    for (int r = 0; r < 8; ++r) {
      const float v = fast_tanh(c0[r] + c1[r] + (float)pcur[r]);
      hout[(khalf * 8 + r) * HP + wave * 16 + col16] = (_Float16)v;
    }
    __syncthreads();
  }

  if (wave < 4) rnn_out_tail(&hbuf[0][0], Wout, bout, out, b0, wave, lane);
}

// ---------------------------------------------------------------------------
// Fallback: fused recurrence (no workspace). x tile pipelined global->reg->LDS.
// ---------------------------------------------------------------------------
__global__ __launch_bounds__(512) void rnn_fused_kernel(
    const float* __restrict__ x, const float* __restrict__ Wxs,
    const float* __restrict__ bxs, const float* __restrict__ Wss,
    const float* __restrict__ bss, const float* __restrict__ Wout,
    const float* __restrict__ bout, float* __restrict__ out) {
  __shared__ _Float16 hbuf[2][16 * HP];
  __shared__ float    xbuf[2][16 * XPF];

  const int tid = threadIdx.x;
  const int wave = tid >> 5, lane = tid & 31;
  const int btile = blockIdx.x;
  const int b0 = btile * 16;
  const int col16 = lane & 15, khalf = lane >> 4;
  const int ncol = wave * 16 + col16;

  v16h bx[4], bs[8];
#pragma unroll
  for (int kk = 0; kk < 4; ++kk) bx[kk] = load_b_w(Wxs, HH, lane, kk * 32, wave * 16);
#pragma unroll
  for (int kk = 0; kk < 8; ++kk) bs[kk] = load_b_w(Wss, HH, lane, kk * 32, wave * 16);
  const float bias = bxs[ncol] + bss[ncol];

  {
    unsigned* p = (unsigned*)&hbuf[0][0];
    for (int i = tid; i < 16 * HP / 2; i += 512) p[i] = 0u;
  }
  const int xr = tid >> 5, xc = (tid & 31) * 4;  // 512 thr * 4 f32 = 2048
  {
    v4f g = *(const v4f*)(x + ((size_t)(b0 + xr) * TT + 0) * II + xc);
    *(v4f*)(&xbuf[0][xr * XPF + xc]) = g;
  }
  __syncthreads();

  for (int t = 0; t < TT; ++t) {
    v4f gx = {};
    if (t + 1 < TT)  // prefetch next time step into registers (latency hidden)
      gx = *(const v4f*)(x + ((size_t)(b0 + xr) * TT + (t + 1)) * II + xc);

    const float*    xin = &xbuf[t & 1][0];
    const _Float16* hin = &hbuf[t & 1][0];
    v8f c0 = {}, c1 = {};
#pragma unroll
    for (int kk = 0; kk < 4; ++kk) {  // x @ W_xs  (A from f32 LDS, cvt to f16)
      const float* p = xin + col16 * XPF + kk * 32 + khalf * 8;
      v4f f0 = *(const v4f*)p, f1 = *(const v4f*)(p + 4);
      v4f f2 = *(const v4f*)(p + 16), f3 = *(const v4f*)(p + 20);
      v16h a;
#pragma unroll
      for (int j = 0; j < 4; ++j) {
        a[j] = (_Float16)f0[j];      a[4 + j]  = (_Float16)f1[j];
        a[8 + j] = (_Float16)f2[j];  a[12 + j] = (_Float16)f3[j];
      }
      if (kk & 1) c1 = WMMA_F16(a, bx[kk], c1);
      else        c0 = WMMA_F16(a, bx[kk], c0);
    }
#pragma unroll
    for (int kk = 0; kk < 8; ++kk) {  // h @ W_ss
      v16h a = load_a_f16(hin, HP, lane, kk * 32);
      if (kk & 1) c1 = WMMA_F16(a, bs[kk], c1);
      else        c0 = WMMA_F16(a, bs[kk], c0);
    }

    _Float16* hout = &hbuf[(t + 1) & 1][0];
#pragma unroll
    for (int r = 0; r < 8; ++r) {
      const float v = fast_tanh(c0[r] + c1[r] + bias);
      hout[(khalf * 8 + r) * HP + ncol] = (_Float16)v;
    }
    if (t + 1 < TT) *(v4f*)(&xbuf[(t + 1) & 1][xr * XPF + xc]) = gx;
    __syncthreads();
  }

  if (wave < 4) rnn_out_tail(&hbuf[0][0], Wout, bout, out, b0, wave, lane);
}

extern "C" void kernel_launch(void* const* d_in, const int* in_sizes, int n_in,
                              void* d_out, int out_size, void* d_ws, size_t ws_size,
                              hipStream_t stream) {
  (void)in_sizes; (void)n_in; (void)out_size;
  const float* x    = (const float*)d_in[0];
  const float* Wxs  = (const float*)d_in[1];
  const float* bxs  = (const float*)d_in[2];
  const float* Wss  = (const float*)d_in[3];
  const float* bss  = (const float*)d_in[4];
  const float* Wout = (const float*)d_in[5];
  const float* bout = (const float*)d_in[6];
  float* out = (float*)d_out;

  const size_t need = (size_t)TT * BB * HH * 2;  // 128 MB of f16 `pre`
  if (ws_size >= need) {
    _Float16* pre = (_Float16*)d_ws;
    rnn_pre_kernel<<<dim3(8, 256), 256, 0, stream>>>(x, Wxs, bxs, bss, pre);
    rnn_rec_kernel<<<dim3(8), 512, 0, stream>>>(pre, Wss, Wout, bout, out);
  } else {
    rnn_fused_kernel<<<dim3(8), 512, 0, stream>>>(x, Wxs, bxs, Wss, bss, Wout, bout, out);
  }
}
